// ParetoSetModel_Transformer_5265629905217
// MI455X (gfx1250) — compile-verified, matrix operand order checked by reading, not run on hardware
//
#include <hip/hip_runtime.h>
#include <hip/hip_bf16.h>
#include <math.h>

#define HIDDEN 256
#define NOBJ   3
#define NDIM   128
#define BPW    16              // batch elements per workgroup
#define ROWS   (BPW * NOBJ)    // 48 rows (3 M-tiles of 16)
#define LSTR   264             // LDS row stride (bf16 elems), pad to dodge bank conflicts
#define KT     8               // K tiles: 256 / 32

typedef __attribute__((ext_vector_type(16))) __bf16 v16bf;
typedef __attribute__((ext_vector_type(8)))  __bf16 v8bf;
typedef __attribute__((ext_vector_type(8)))  float  v8f;

// Load a 16x32 bf16 fragment (A or B operand) for V_WMMA_F32_16X16X32_BF16.
// ISA 7.12.2 layout: lane(0-15)=row r, VGPR0-3 hold K = h*8 + 0..7, VGPR4-7 hold
// K = 16 + h*8 + 0..7 where h = lane>>4.  Caller passes p = base + row*stride + k0 + h*8.
__device__ __forceinline__ v16bf load_frag16x32(const __bf16* p) {
  v8bf lo = *(const v8bf*)(p);
  v8bf hi = *(const v8bf*)(p + 16);
  return __builtin_shufflevector(lo, hi, 0,1,2,3,4,5,6,7,8,9,10,11,12,13,14,15);
}

// 48xK(=256) x Kx(NTPW*8*16) GEMM: 3 M-tiles, NTPW N-tiles per wave, 8 waves.
// A: bf16 row-major in LDS (stride ast).  W: bf16 row-major [N][256] in global (L2-hot).
// epi(row, col, val) consumes each f32 result element.
template <int NTPW, typename Epi>
__device__ __forceinline__ void gemm3xN(const __bf16* __restrict__ A, int ast,
                                        const __bf16* __restrict__ W, Epi epi) {
  const int wave = threadIdx.x >> 5;
  const int lane = threadIdx.x & 31;
  const int r = lane & 15;
  const int h = lane >> 4;
#pragma unroll
  for (int m = 0; m < 3; ++m) {
    v16bf a[KT];                       // cache A-frags for all K-steps (reused across N-tiles)
#pragma unroll
    for (int ks = 0; ks < KT; ++ks)
      a[ks] = load_frag16x32(A + (m * 16 + r) * ast + ks * 32 + h * 8);
#pragma unroll
    for (int t = 0; t < NTPW; ++t) {
      const int n0 = (wave * NTPW + t) * 16;
      v8f acc = {};
#pragma unroll
      for (int ks = 0; ks < KT; ++ks) {
        v16bf b = load_frag16x32(W + (n0 + r) * HIDDEN + ks * 32 + h * 8);
        acc = __builtin_amdgcn_wmma_f32_16x16x32_bf16(false, a[ks], false, b,
                                                      (short)0, acc, false, false);
      }
      // C/D layout: VGPR v, lanes 0-15 -> M=v, lanes 16-31 -> M=v+8; N = lane&15.
#pragma unroll
      for (int v = 0; v < 8; ++v)
        epi(m * 16 + v + h * 8, n0 + r, acc[v]);
    }
  }
}

__global__ void conv_bf16(const float* __restrict__ src, __bf16* __restrict__ dst, int n) {
  int i = blockIdx.x * blockDim.x + threadIdx.x;
  if (i < n) dst[i] = (__bf16)src[i];
}

__global__ __launch_bounds__(256, 1)   // LDS-limited occupancy; let backend use full VGPR file
void pareto_fused(const float* __restrict__ ray,
                  const float* __restrict__ emb_w, const float* __restrict__ emb_b,
                  const float* __restrict__ in_proj_b, const float* __restrict__ out_proj_b,
                  const float* __restrict__ ffn1_b, const float* __restrict__ ffn2_b,
                  const float* __restrict__ out_b,
                  const __bf16* __restrict__ w_qkv, const __bf16* __restrict__ w_op,
                  const __bf16* __restrict__ w_f1, const __bf16* __restrict__ w_f2,
                  const __bf16* __restrict__ w_out,
                  float* __restrict__ out) {
  extern __shared__ char smem_raw[];
  __bf16* xb = (__bf16*)smem_raw;          // [48][264] activations x
  __bf16* qb = xb + ROWS * LSTR;           // q, later ctx   (qb,kb,vb contiguous!)
  __bf16* kb = qb + ROWS * LSTR;           // k, later ffn hidden, later y (f32)
  __bf16* vb = kb + ROWS * LSTR;           // v, later x after ffn
  float*  sc = (float*)(vb + ROWS * LSTR); // [16][3][3] scores / attn weights

  const int tid = threadIdx.x;
  const int b0  = blockIdx.x * BPW;

  // ---- Stage A: x = relu(ray * emb_w + emb_b), rows = lb*3 + l ----
  for (int idx = tid; idx < ROWS * HIDDEN; idx += 256) {
    int row = idx >> 8, k = idx & 255;
    int lb = row / 3, l = row - lb * 3;
    float rv = ray[(b0 + lb) * NOBJ + l];
    float v = rv * emb_w[l * HIDDEN + k] + emb_b[l * HIDDEN + k];
    xb[row * LSTR + k] = (__bf16)fmaxf(v, 0.f);
  }
  __syncthreads();

  // ---- Stage B: qkv = x @ in_proj_w^T + b  (N = 768, 6 N-tiles/wave) ----
  // qb/kb/vb are contiguous: select the destination slab with integer math,
  // NOT a pointer select (pointer selects spilled to scratch in earlier builds).
  gemm3xN<6>(xb, LSTR, w_qkv, [=](int row, int col, float v) {
    float val = v + in_proj_b[col];
    qb[(col >> 8) * (ROWS * LSTR) + row * LSTR + (col & (HIDDEN - 1))] = (__bf16)val;
  });
  __syncthreads();

  // ---- Attention scores: 16 batches x 3x3 dot products of length 256 (VALU) ----
  if (tid < BPW * 9) {
    int lb = tid / 9, rem = tid - lb * 9;
    int l = rem / 3, mm = rem - l * 3;
    const __bf16* qr = qb + (lb * 3 + l) * LSTR;
    const __bf16* kr = kb + (lb * 3 + mm) * LSTR;
    float d = 0.f;
    for (int k = 0; k < HIDDEN; ++k) d += (float)qr[k] * (float)kr[k];
    sc[tid] = d * 0.0625f;               // * 1/sqrt(256)
  }
  __syncthreads();
  // softmax over rows of 3
  if (tid < ROWS) {
    float* s = sc + tid * 3;
    float m  = fmaxf(s[0], fmaxf(s[1], s[2]));
    float e0 = expf(s[0] - m), e1 = expf(s[1] - m), e2 = expf(s[2] - m);
    float inv = 1.f / (e0 + e1 + e2);
    s[0] = e0 * inv; s[1] = e1 * inv; s[2] = e2 * inv;
  }
  __syncthreads();
  // ctx = attn @ v  -> overwrite qb
  for (int idx = tid; idx < ROWS * HIDDEN; idx += 256) {
    int row = idx >> 8, c = idx & 255;
    int lb = row / 3;
    const float* a = sc + row * 3;
    float v = a[0] * (float)vb[(lb * 3 + 0) * LSTR + c]
            + a[1] * (float)vb[(lb * 3 + 1) * LSTR + c]
            + a[2] * (float)vb[(lb * 3 + 2) * LSTR + c];
    qb[row * LSTR + c] = (__bf16)v;
  }
  __syncthreads();

  // ---- Stage D: x = x + (ctx @ out_proj_w^T + b)  -> xb (element owned by one lane) ----
  gemm3xN<2>(qb, LSTR, w_op, [=](int row, int col, float v) {
    int i = row * LSTR + col;
    xb[i] = (__bf16)(v + out_proj_b[col] + (float)xb[i]);
  });
  __syncthreads();

  // ---- Stage E: h = relu(x @ ffn1_w^T + b) -> kb ----
  gemm3xN<2>(xb, LSTR, w_f1, [=](int row, int col, float v) {
    kb[row * LSTR + col] = (__bf16)fmaxf(v + ffn1_b[col], 0.f);
  });
  __syncthreads();

  // ---- Stage F: x = x + (h @ ffn2_w^T + b) -> vb (can't write xb: it's A elsewhere) ----
  gemm3xN<2>(kb, LSTR, w_f2, [=](int row, int col, float v) {
    vb[row * LSTR + col] = (__bf16)(v + ffn2_b[col] + (float)xb[row * LSTR + col]);
  });
  __syncthreads();

  // ---- Stage G: y = x @ out_w^T + b  (N = 128, 1 N-tile/wave) -> f32 in kb region ----
  float* yb = (float*)kb;                  // [48][132] f32, exactly fits kb's 25344 B
  gemm3xN<1>(vb, LSTR, w_out, [=](int row, int col, float v) {
    yb[row * 132 + col] = v + out_b[col];
  });
  __syncthreads();

  // ---- Stage H: mean over tokens, scaled sigmoid, write out ----
  for (int idx = tid; idx < BPW * NDIM; idx += 256) {
    int lb = idx >> 7, d = idx & 127;
    float s = (yb[(lb * 3 + 0) * 132 + d] + yb[(lb * 3 + 1) * 132 + d]
             + yb[(lb * 3 + 2) * 132 + d]) * (1.f / 3.f);
    float sig = 1.f / (1.f + expf(-s));
    out[(size_t)(b0 + lb) * NDIM + d] = 2.f * sig - 1.f;
  }
}

extern "C" void kernel_launch(void* const* d_in, const int* in_sizes, int n_in,
                              void* d_out, int out_size, void* d_ws, size_t ws_size,
                              hipStream_t stream) {
  const float* ray        = (const float*)d_in[0];
  const float* emb_w      = (const float*)d_in[1];
  const float* emb_b      = (const float*)d_in[2];
  const float* in_proj_w  = (const float*)d_in[3];
  const float* in_proj_b  = (const float*)d_in[4];
  const float* out_proj_w = (const float*)d_in[5];
  const float* out_proj_b = (const float*)d_in[6];
  const float* ffn1_w     = (const float*)d_in[7];
  const float* ffn1_b     = (const float*)d_in[8];
  const float* ffn2_w     = (const float*)d_in[9];
  const float* ffn2_b     = (const float*)d_in[10];
  const float* out_w      = (const float*)d_in[11];
  const float* out_b      = (const float*)d_in[12];

  // bf16 weight staging in workspace (row-major [out_features][256], ~850 KB, L2-resident)
  __bf16* w_qkv = (__bf16*)d_ws;
  __bf16* w_op  = w_qkv + 768 * 256;
  __bf16* w_f1  = w_op  + 256 * 256;
  __bf16* w_f2  = w_f1  + 256 * 256;
  __bf16* w_out = w_f2  + 256 * 256;

  conv_bf16<<<(768 * 256 + 255) / 256, 256, 0, stream>>>(in_proj_w,  w_qkv, 768 * 256);
  conv_bf16<<<(256 * 256 + 255) / 256, 256, 0, stream>>>(out_proj_w, w_op,  256 * 256);
  conv_bf16<<<(256 * 256 + 255) / 256, 256, 0, stream>>>(ffn1_w,     w_f1,  256 * 256);
  conv_bf16<<<(256 * 256 + 255) / 256, 256, 0, stream>>>(ffn2_w,     w_f2,  256 * 256);
  conv_bf16<<<(128 * 256 + 255) / 256, 256, 0, stream>>>(out_w,      w_out, 128 * 256);

  size_t smem = (size_t)4 * ROWS * LSTR * sizeof(__bf16) + BPW * 9 * sizeof(float);
  hipFuncSetAttribute((const void*)pareto_fused,
                      hipFuncAttributeMaxDynamicSharedMemorySize, (int)smem);
  pareto_fused<<<65536 / BPW, 256, smem, stream>>>(
      ray, emb_w, emb_b, in_proj_b, out_proj_b, ffn1_b, ffn2_b, out_b,
      w_qkv, w_op, w_f1, w_f2, w_out, (float*)d_out);
}
